// Conv1dMultiHeadAttention_8108898254870
// MI455X (gfx1250) — compile-verified, matrix operand order checked by reading, not run on hardware
//
#include <hip/hip_runtime.h>

// ---------------------------------------------------------------------------
// Types for CDNA5 WMMA (gfx1250, wave32)
// ---------------------------------------------------------------------------
typedef __bf16 bf16_t;
typedef __bf16 v16bf __attribute__((ext_vector_type(16)));
typedef __bf16 v8bf  __attribute__((ext_vector_type(8)));
typedef __bf16 v4bf  __attribute__((ext_vector_type(4)));
typedef float  v8f   __attribute__((ext_vector_type(8)));
typedef unsigned u32x4 __attribute__((ext_vector_type(4)));
typedef unsigned u32x8 __attribute__((ext_vector_type(8)));

#define WMMA_BF16(A, B, C) \
  __builtin_amdgcn_wmma_f32_16x16x32_bf16(false, (A), false, (B), (short)0, (C), false, false)

static __device__ __forceinline__ v16bf join8(v8bf lo, v8bf hi) {
  return __builtin_shufflevector(lo, hi, 0,1,2,3,4,5,6,7,8,9,10,11,12,13,14,15);
}

static __device__ __forceinline__ bf16_t f2bf(float f) {
  unsigned u = __builtin_bit_cast(unsigned, f);
  unsigned r = (u + 0x7FFFu + ((u >> 16) & 1u)) >> 16;
  unsigned short h = (unsigned short)r;
  return __builtin_bit_cast(bf16_t, h);
}

static __device__ __forceinline__ unsigned rfl(unsigned x) {
  return (unsigned)__builtin_amdgcn_readfirstlane((int)x);
}

// Problem constants
#define Bc   2
#define Sc   2048
#define DINc 1024
#define Hc   16
#define DAc  64
#define DHc  64
#define EMBc 1024
#define Mrows (Bc * Sc)   // 4096

// ---------------------------------------------------------------------------
// f32 -> bf16 conversion (x4 vectorized)
// ---------------------------------------------------------------------------
__global__ __launch_bounds__(256)
void f32_to_bf16_x4(const float* __restrict__ in, bf16_t* __restrict__ out, int n4) {
  int i = blockIdx.x * blockDim.x + threadIdx.x;
  if (i < n4) {
    float4 f = ((const float4*)in)[i];
    v4bf o;
    o[0] = f2bf(f.x); o[1] = f2bf(f.y); o[2] = f2bf(f.z); o[3] = f2bf(f.w);
    ((v4bf*)out)[i] = o;
  }
}

// ---------------------------------------------------------------------------
// WMMA GEMM: O(M x N) = A(M x 1024) * W(N x 1024)^T
//   mode 0: write bf16 as (B,H,S,DA)        (Q and K)
//   mode 1: write bf16 as (B,H,DH,S)        (V transposed)
//   mode 2: write f32  as (M x N) + bias    (final projection)
// Tile BM=128, BN=64, BK=64; 256 threads = 8 waves; wave -> 32x32 subtile.
// A tile: GLOBAL_LOAD_ASYNC_TO_LDS_B128 (4 instr/wave/stage, ASYNCcnt-gated).
// W tile: one TENSOR_LOAD_TO_LDS (TDM) per stage, issued by wave 0 only,
//         with D# padding reproducing the 72-bf16 LDS row stride,
//         TENSORcnt-gated. Both double buffered.
// ---------------------------------------------------------------------------
#define BM   128
#define BN   64
#define BKS  64
#define APAD 72   // LDS row stride (bf16): 36 dwords -> conflict-free b128 reads

__global__ __launch_bounds__(256)
void wmma_gemm_kernel(const bf16_t* __restrict__ A, const bf16_t* __restrict__ W,
                      bf16_t* __restrict__ outb, float* __restrict__ outf,
                      const float* __restrict__ bias, int mode) {
  __shared__ alignas(16) bf16_t As[2][BM * APAD];
  __shared__ alignas(16) bf16_t Ws[2][BN * APAD];

  const int tid  = threadIdx.x;
  const int m0   = blockIdx.x * BM;
  const int n0   = blockIdx.y * BN;
  const int wave = tid >> 5;
  const int lane = tid & 31;
  const int wm   = wave & 3;   // 0..3 : M subtile
  const int wn   = wave >> 2;  // 0..1 : N subtile
  const int lr   = lane & 15;
  const int hf   = lane >> 4;
  const int kb   = hf * 8;     // K-half base per ISA A/B layout

  const unsigned long long abase = (unsigned long long)A;
  const unsigned long long wbase = (unsigned long long)W;

  // ---- per-wave async staging of the A tile: 4 chunks / thread / stage
  auto issue_A = [&](int k0, int buf) {
    #pragma unroll
    for (int i = 0; i < 4; ++i) {
      int cid = tid + i * 256;          // 1024 chunks: 128 rows x 8
      int row = cid >> 3, c = cid & 7;
      unsigned lds  = (unsigned)(unsigned long long)(void*)&As[buf][row * APAD + c * 8];
      unsigned voff = (unsigned)((((m0 + row) * DINc) + k0 + c * 8) * 2);
      asm volatile("global_load_async_to_lds_b128 %0, %1, %2"
                   :: "v"(lds), "v"(voff), "s"(abase) : "memory");
    }
  };

  // ---- TDM staging of the W tile: one tensor_load_to_lds per stage (wave 0)
  // D#: 64x64 bf16 tile of a 1024x1024 bf16 row-major tensor, LDS-padded
  // 16B after every 128B row (pad_interval=4 -> 32 dwords, pad_amount=3 -> 4).
  auto issue_W_tdm = [&](int k0, int buf) {
    unsigned ldsw = rfl((unsigned)(unsigned long long)(void*)&Ws[buf][0]);
    unsigned long long ga = wbase + (unsigned long long)(((unsigned)(n0 * DINc + k0)) * 2u);
    u32x4 g0;
    g0[0] = 1u;                                   // count=1 (valid descriptor)
    g0[1] = ldsw;                                 // lds_addr
    g0[2] = rfl((unsigned)ga);                    // global_addr[31:0]
    g0[3] = rfl((unsigned)(ga >> 32)) | 0x80000000u;  // addr[56:32] | type=2
    u32x8 g1;
    g1[0] = (1u << 16) | (1u << 20) | (4u << 22) | (3u << 25);
                 // data_size=2B | pad_enable | pad_interval=32dw | pad_amount=4dw
    g1[1] = (DINc & 0xFFFFu) << 16;               // tensor_dim0 -> bits 79:48
    g1[2] = (DINc >> 16) | ((Mrows & 0xFFFFu) << 16) /* tensor_dim1 lo */;
    g1[2] = (DINc >> 16) | ((DINc & 0xFFFFu) << 16); // tensor_dim1 = 1024
    g1[3] = (DINc >> 16) | (64u << 16);           // tensor_dim1 hi | tile_dim0=64
    g1[4] = 64u;                                  // tile_dim1=64, tile_dim2=0
    g1[5] = (unsigned)DINc;                       // tensor_dim0_stride = 1024
    g1[6] = 0u;
    g1[7] = 0u;
    asm volatile("tensor_load_to_lds %0, %1" :: "s"(g0), "s"(g1) : "memory");
  };

  issue_A(0, 0);
  if (wave == 0) issue_W_tdm(0, 0);

  v8f acc[2][2] = {};

  for (int k0 = 0; k0 < DINc; k0 += BKS) {
    const int buf  = (k0 / BKS) & 1;
    const bool more = (k0 + BKS) < DINc;
    __syncthreads();  // all waves finished reading buf^1 (previous stage)
    if (more) issue_A(k0 + BKS, buf ^ 1);
    if (wave == 0) {
      if (more) {
        issue_W_tdm(k0 + BKS, buf ^ 1);
        __builtin_amdgcn_s_wait_tensorcnt(1);   // current stage W landed
      } else {
        __builtin_amdgcn_s_wait_tensorcnt(0);
      }
    }
    if (more) asm volatile("s_wait_asynccnt 0x4" ::: "memory");  // stage A landed
    else      asm volatile("s_wait_asynccnt 0x0" ::: "memory");
    __syncthreads();  // stage data visible to all waves

    #pragma unroll
    for (int kk = 0; kk < 2; ++kk) {
      const int cb = kk * 32;
      v16bf af[2], wf[2];
      #pragma unroll
      for (int t = 0; t < 2; ++t) {
        int ar = (wm * 32 + t * 16 + lr) * APAD + cb;
        af[t] = join8(*(const v8bf*)&As[buf][ar + kb], *(const v8bf*)&As[buf][ar + 16 + kb]);
        int wr = (wn * 32 + t * 16 + lr) * APAD + cb;
        wf[t] = join8(*(const v8bf*)&Ws[buf][wr + kb], *(const v8bf*)&Ws[buf][wr + 16 + kb]);
      }
      #pragma unroll
      for (int tm = 0; tm < 2; ++tm)
        #pragma unroll
        for (int tn = 0; tn < 2; ++tn)
          acc[tm][tn] = WMMA_BF16(af[tm], wf[tn], acc[tm][tn]);
    }
  }

  // Epilogue (mode hoisted). C layout: VGPR r -> row (r + 8*hf), col lr.
  const int mrow0 = m0 + wm * 32;
  const int ncol0 = n0 + wn * 32;
  if (mode == 2) {
    #pragma unroll
    for (int tm = 0; tm < 2; ++tm)
      #pragma unroll
      for (int tn = 0; tn < 2; ++tn) {
        int n = ncol0 + tn * 16 + lr;
        float bv = bias[n];
        #pragma unroll
        for (int r = 0; r < 8; ++r) {
          int sflat = mrow0 + tm * 16 + r + 8 * hf;
          outf[(size_t)sflat * EMBc + n] = acc[tm][tn][r] + bv;
        }
      }
  } else if (mode == 0) {
    #pragma unroll
    for (int tm = 0; tm < 2; ++tm)
      #pragma unroll
      for (int tn = 0; tn < 2; ++tn) {
        int n = ncol0 + tn * 16 + lr;
        int h = n >> 6, d = n & 63;
        #pragma unroll
        for (int r = 0; r < 8; ++r) {
          int sflat = mrow0 + tm * 16 + r + 8 * hf;
          int b = sflat >> 11, s = sflat & (Sc - 1);
          outb[(((size_t)(b * Hc + h) * Sc) + s) * DAc + d] = f2bf(acc[tm][tn][r]);
        }
      }
  } else {
    #pragma unroll
    for (int tm = 0; tm < 2; ++tm)
      #pragma unroll
      for (int tn = 0; tn < 2; ++tn) {
        int n = ncol0 + tn * 16 + lr;
        int h = n >> 6, d = n & 63;
        #pragma unroll
        for (int r = 0; r < 8; ++r) {
          int sflat = mrow0 + tm * 16 + r + 8 * hf;
          int b = sflat >> 11, s = sflat & (Sc - 1);
          outb[(((size_t)(b * Hc + h) * DHc) + d) * Sc + s] = f2bf(acc[tm][tn][r]);
        }
      }
  }
}

// ---------------------------------------------------------------------------
// Causal flash attention, one wave per 16 query rows, WMMA throughout.
//   Q,K: (B,H,S,64) bf16; Vt: (B,H,64,S) bf16; Ctx out: (B,S,H*64) bf16
// ---------------------------------------------------------------------------
#define LPAD 40

__global__ __launch_bounds__(128)
void attn_kernel(const bf16_t* __restrict__ Q, const bf16_t* __restrict__ K,
                 const bf16_t* __restrict__ Vt, bf16_t* __restrict__ Ctx) {
  __shared__ alignas(16) bf16_t P[4][16 * LPAD];  // per-wave P tile (16x32, padded)

  const int lane = threadIdx.x & 31;
  const int wave = threadIdx.x >> 5;
  const int lr   = lane & 15;
  const int hf   = lane >> 4;
  const int kb   = hf * 8;
  const int bh   = blockIdx.y;                    // b*H + h
  const int q0   = blockIdx.x * 64 + wave * 16;   // this wave's query base

  const bf16_t* qrow = Q + ((size_t)bh * Sc + q0 + lr) * DAc;
  v16bf qf0 = join8(*(const v8bf*)(qrow + kb),      *(const v8bf*)(qrow + 16 + kb));
  v16bf qf1 = join8(*(const v8bf*)(qrow + 32 + kb), *(const v8bf*)(qrow + 48 + kb));

  v8f o[4] = {};
  float mrow[8], lrow[8];
  #pragma unroll
  for (int r = 0; r < 8; ++r) { mrow[r] = -1e30f; lrow[r] = 0.0f; }
  const float scale = 0.125f;  // DH^-0.5

  const int jhi = q0 + 16;  // keys needed: 0 .. q0+15 (causal)
  for (int j0 = 0; j0 < jhi; j0 += 32) {
    const bf16_t* k0p = K + ((size_t)bh * Sc + j0 + lr) * DAc;
    __builtin_prefetch((const void*)(k0p + 32 * DAc), 0, 3);
    __builtin_prefetch((const void*)(Vt + ((size_t)bh * DHc + lr) * Sc + j0 + 32), 0, 3);

    // ---- scores: 16 queries x 32 keys
    v8f s0 = {}, s1 = {};
    {
      v16bf kf0 = join8(*(const v8bf*)(k0p + kb),      *(const v8bf*)(k0p + 16 + kb));
      v16bf kf1 = join8(*(const v8bf*)(k0p + 32 + kb), *(const v8bf*)(k0p + 48 + kb));
      s0 = WMMA_BF16(qf0, kf0, s0);
      s0 = WMMA_BF16(qf1, kf1, s0);
      const bf16_t* k1p = k0p + 16 * DAc;
      v16bf kg0 = join8(*(const v8bf*)(k1p + kb),      *(const v8bf*)(k1p + 16 + kb));
      v16bf kg1 = join8(*(const v8bf*)(k1p + 32 + kb), *(const v8bf*)(k1p + 48 + kb));
      s1 = WMMA_BF16(qf0, kg0, s1);
      s1 = WMMA_BF16(qf1, kg1, s1);
    }

    // ---- mask + online softmax (per C-layout row r, half hf)
    #pragma unroll
    for (int r = 0; r < 8; ++r) {
      int qi  = q0 + r + 8 * hf;
      int ka  = j0 + lr;
      int kbi = j0 + 16 + lr;
      float a = s0[r] * scale;
      float b = s1[r] * scale;
      if (ka  > qi) a = -1e30f;
      if (kbi > qi) b = -1e30f;
      float mx = fmaxf(a, b);
      #pragma unroll
      for (int d = 1; d < 16; d <<= 1) mx = fmaxf(mx, __shfl_xor(mx, d, 32));
      float mnew = fmaxf(mrow[r], mx);
      float pa = __expf(a - mnew);
      float pb = __expf(b - mnew);
      float rs = pa + pb;
      #pragma unroll
      for (int d = 1; d < 16; d <<= 1) rs += __shfl_xor(rs, d, 32);
      float alpha = __expf(mrow[r] - mnew);
      lrow[r] = lrow[r] * alpha + rs;
      mrow[r] = mnew;
      #pragma unroll
      for (int t = 0; t < 4; ++t) o[t][r] *= alpha;
      P[wave][(r + 8 * hf) * LPAD + lr]      = f2bf(pa);
      P[wave][(r + 8 * hf) * LPAD + 16 + lr] = f2bf(pb);
    }
    asm volatile("s_wait_dscnt 0x0" ::: "memory");  // intra-wave LDS RAW

    v16bf pf;
    {
      const bf16_t* pr = &P[wave][lr * LPAD];
      pf = join8(*(const v8bf*)(pr + kb), *(const v8bf*)(pr + 16 + kb));
    }
    asm volatile("" ::: "memory");

    // ---- context: P(16x32) x V(32x64), B-fragments from V^T rows
    #pragma unroll
    for (int t = 0; t < 4; ++t) {
      const bf16_t* vrow = Vt + ((size_t)bh * DHc + t * 16 + lr) * Sc + j0;
      v16bf vf = join8(*(const v8bf*)(vrow + kb), *(const v8bf*)(vrow + 16 + kb));
      o[t] = WMMA_BF16(pf, vf, o[t]);
    }
    asm volatile("s_wait_dscnt 0x0" ::: "memory");  // WAR on P before next iter
  }

  // ---- normalize + scatter to Ctx (B,S,H*DH) bf16
  const int b = bh >> 4, h = bh & 15;
  #pragma unroll
  for (int r = 0; r < 8; ++r) {
    float inv = 1.0f / lrow[r];
    int s = q0 + r + 8 * hf;
    bf16_t* crow = Ctx + ((size_t)(b * Sc + s)) * (Hc * DHc) + h * DHc;
    #pragma unroll
    for (int t = 0; t < 4; ++t)
      crow[t * 16 + lr] = f2bf(o[t][r] * inv);
  }
}

// ---------------------------------------------------------------------------
// Launch
// ---------------------------------------------------------------------------
extern "C" void kernel_launch(void* const* d_in, const int* in_sizes, int n_in,
                              void* d_out, int out_size, void* d_ws, size_t ws_size,
                              hipStream_t stream) {
  const float* x     = (const float*)d_in[0];
  // d_in[1] = attn_mask (causal tril) — implied by kernel, unused
  const float* wq    = (const float*)d_in[2];
  const float* wk    = (const float*)d_in[3];
  const float* wv    = (const float*)d_in[4];
  const float* w_out = (const float*)d_in[5];
  const float* b_out = (const float*)d_in[6];
  float* out = (float*)d_out;

  char* ws = (char*)d_ws;
  const size_t XB = (size_t)Mrows * DINc;   // 4,194,304 elems
  const size_t WB = (size_t)DINc * DINc;    // 1,048,576 elems
  bf16_t* Xb  = (bf16_t*)(ws);              // 8 MB
  bf16_t* Wqb = (bf16_t*)(ws + 8388608);    // 2 MB
  bf16_t* Wkb = (bf16_t*)(ws + 10485760);   // 2 MB
  bf16_t* Wvb = (bf16_t*)(ws + 12582912);   // 2 MB
  bf16_t* Wob = (bf16_t*)(ws + 14680064);   // 2 MB
  bf16_t* Qb  = (bf16_t*)(ws + 16777216);   // 8 MB
  bf16_t* Kb  = (bf16_t*)(ws + 25165824);   // 8 MB
  bf16_t* Vtb = (bf16_t*)(ws + 33554432);   // 8 MB
  bf16_t* Ctx = (bf16_t*)(ws + 41943040);   // 8 MB  (total 48 MB)

  // 1) convert to bf16
  f32_to_bf16_x4<<<dim3((unsigned)(XB / 4 / 256)), dim3(256), 0, stream>>>(x, Xb, (int)(XB / 4));
  f32_to_bf16_x4<<<dim3((unsigned)(WB / 4 / 256)), dim3(256), 0, stream>>>(wq, Wqb, (int)(WB / 4));
  f32_to_bf16_x4<<<dim3((unsigned)(WB / 4 / 256)), dim3(256), 0, stream>>>(wk, Wkb, (int)(WB / 4));
  f32_to_bf16_x4<<<dim3((unsigned)(WB / 4 / 256)), dim3(256), 0, stream>>>(wv, Wvb, (int)(WB / 4));
  f32_to_bf16_x4<<<dim3((unsigned)(WB / 4 / 256)), dim3(256), 0, stream>>>(w_out, Wob, (int)(WB / 4));

  // 2) QKV projections (WMMA GEMM, async-LDS + TDM staged, double buffered)
  dim3 ggrid(Mrows / BM, DINc / BN);  // 32 x 16
  wmma_gemm_kernel<<<ggrid, dim3(256), 0, stream>>>(Xb, Wqb, Qb,  nullptr, nullptr, 0);
  wmma_gemm_kernel<<<ggrid, dim3(256), 0, stream>>>(Xb, Wkb, Kb,  nullptr, nullptr, 0);
  wmma_gemm_kernel<<<ggrid, dim3(256), 0, stream>>>(Xb, Wvb, Vtb, nullptr, nullptr, 1);

  // 3) causal flash attention (WMMA)
  attn_kernel<<<dim3(Sc / 64, Bc * Hc), dim3(128), 0, stream>>>(Qb, Kb, Vtb, Ctx);

  // 4) output projection + bias (WMMA GEMM, f32 out)
  wmma_gemm_kernel<<<ggrid, dim3(256), 0, stream>>>(Ctx, Wob, nullptr, out, b_out, 2);

  (void)in_sizes; (void)n_in; (void)out_size; (void)ws_size;
}
